// Net_17669495456403
// MI455X (gfx1250) — matrix operand order repaired and checked by reading, compiled, not run on hardware
//
#include <hip/hip_runtime.h>
#include <math.h>

// ---------- types ----------
typedef __bf16  bf16_t;
typedef __bf16  v16bf __attribute__((ext_vector_type(16)));
typedef __bf16  v8bf  __attribute__((ext_vector_type(8)));
typedef float   v8f   __attribute__((ext_vector_type(8)));

static __device__ __forceinline__ unsigned short f2bfbits(float f) {
    union { float f; unsigned u; } in; in.f = f;
    unsigned u = in.u;
    return (unsigned short)((u + 0x7FFFu + ((u >> 16) & 1u)) >> 16);   // RNE
}
static __device__ __forceinline__ bf16_t f2bf(float f) {
    union { unsigned short s; bf16_t b; } o; o.s = f2bfbits(f);
    return o.b;
}

// A-fragment (16x32 bf16) from row-major bf16: elems 0-7 at p, 8-15 at p+16.
static __device__ __forceinline__ v16bf load_abf(const bf16_t* __restrict__ p) {
    v8bf lo = *(const v8bf*)(p);
    v8bf hi = *(const v8bf*)(p + 16);
    return __builtin_shufflevector(lo, hi, 0,1,2,3,4,5,6,7,8,9,10,11,12,13,14,15);
}

// ---------- fp32 -> bf16 bulk convert (4 elems/thread) ----------
__global__ void cvt_f32_bf16_kernel(const float* __restrict__ in,
                                    bf16_t* __restrict__ out, unsigned n4) {
    unsigned t = blockIdx.x * blockDim.x + threadIdx.x;
    if (t >= n4) return;
    float4 v = *(const float4*)(in + (size_t)t * 4);
    ushort4 o;
    o.x = f2bfbits(v.x); o.y = f2bfbits(v.y);
    o.z = f2bfbits(v.z); o.w = f2bfbits(v.w);
    *(ushort4*)((unsigned short*)out + (size_t)t * 4) = o;
}

// ---------- weight packing: fp32 [K,N] row-major -> bf16 B-fragment tiles ----------
// Tile (tk,tn) = 32x16 stored as 32 lanes x 16 elems (1KB).
// element (lane l, i): K = tk*32 + (i<8 ? (l>>4)*8 + i : 16 + (l>>4)*8 + i-8), N = tn*16 + (l&15)
__global__ void pack_w_kernel(const float* __restrict__ W, bf16_t* __restrict__ out,
                              unsigned K, unsigned Nout) {
    unsigned t = blockIdx.x * blockDim.x + threadIdx.x;
    unsigned total = K * Nout;
    if (t >= total) return;
    unsigned tile   = t >> 9;
    unsigned within = t & 511;
    unsigned l = within >> 4;
    unsigned i = within & 15;
    unsigned ntiles = Nout >> 4;
    unsigned tk = tile / ntiles, tn = tile % ntiles;
    unsigned k = tk * 32 + (i < 8 ? (l >> 4) * 8 + i : 16 + (l >> 4) * 8 + (i - 8));
    unsigned n = tn * 16 + (l & 15);
    out[t] = f2bf(W[(size_t)k * Nout + n]);
}

// ---------- edge scatter: agg[dst] += x[src] * w (fp32, L2 atomics) ----------
__global__ void scatter_edges_kernel(const float* __restrict__ X,
                                     const int* __restrict__ src,
                                     const int* __restrict__ dst,
                                     const float* __restrict__ w,
                                     float* __restrict__ agg, unsigned E, unsigned F) {
    unsigned t = blockIdx.x * blockDim.x + threadIdx.x;
    unsigned fq = F >> 2;
    unsigned total = E * fq;
    if (t >= total) return;
    unsigned e = t / fq;
    unsigned c = (t % fq) << 2;
    int s = src[e], d = dst[e];
    float ww = w[e];
    float4 xv = *(const float4*)(X + (size_t)s * F + c);
    float* ap = agg + (size_t)d * F + c;
    atomicAdd(ap + 0, xv.x * ww);
    atomicAdd(ap + 1, xv.y * ww);
    atomicAdd(ap + 2, xv.z * ww);
    atomicAdd(ap + 3, xv.w * ww);
}

// ---------- WMMA GEMM: D = A0@B0 (+ A1@B1) + bias, optional relu ----------
// A: bf16 row-major [M,K]; B: packed bf16 fragment tiles.
// Block = 128 threads (4 waves); wave tile 32x64 (2 row-frags x 4 col-frags);
// block tile 128x64. Dual/single K-loops are specialized (uniform branch hoisted)
// so every loop body is one straight-line BB: all fragment loads clause at the
// top and overlap the WMMA chain with graduated s_wait_loadcnt.
__global__ __launch_bounds__(128)
__attribute__((amdgpu_waves_per_eu(1)))
void gemm_bf16_wmma(
        const bf16_t* __restrict__ A0, const bf16_t* __restrict__ B0,
        const bf16_t* __restrict__ A1, const bf16_t* __restrict__ B1,
        const float* __restrict__ bias, float* __restrict__ D,
        int M, int K, int Nout, int relu) {
    const int lane = threadIdx.x & 31;
    const int wave = threadIdx.x >> 5;
    const int m0   = blockIdx.y * 128 + wave * 32;
    const int nb   = blockIdx.x * 64;
    const int r15  = lane & 15;
    const int half = lane >> 4;
    const int ntiles = Nout >> 4;

    int arow0 = m0 + r15;       if (arow0 > M - 1) arow0 = M - 1;
    int arow1 = m0 + 16 + r15;  if (arow1 > M - 1) arow1 = M - 1;
    const bf16_t* a0p0 = A0 + (size_t)arow0 * K + half * 8;
    const bf16_t* a0p1 = A0 + (size_t)arow1 * K + half * 8;

    v8f acc[2][4];
    #pragma unroll
    for (int rt = 0; rt < 2; ++rt)
        #pragma unroll
        for (int j = 0; j < 4; ++j)
            #pragma unroll
            for (int i = 0; i < 8; ++i) acc[rt][j][i] = 0.0f;

    const int kt_end = K >> 5;

    if (A1 != nullptr) {
        const bf16_t* a1p0 = A1 + (size_t)arow0 * K + half * 8;
        const bf16_t* a1p1 = A1 + (size_t)arow1 * K + half * 8;
        for (int kt = 0; kt < kt_end; ++kt) {
            const size_t bbase =
                (((size_t)kt * ntiles + (nb >> 4)) << 9) + (size_t)lane * 16;
            v16bf a00 = load_abf(a0p0 + kt * 32);
            v16bf a01 = load_abf(a0p1 + kt * 32);
            v16bf a10 = load_abf(a1p0 + kt * 32);
            v16bf a11 = load_abf(a1p1 + kt * 32);
            v16bf b0 = *(const v16bf*)(B0 + bbase);
            v16bf b1 = *(const v16bf*)(B0 + bbase + 512);
            v16bf b2 = *(const v16bf*)(B0 + bbase + 1024);
            v16bf b3 = *(const v16bf*)(B0 + bbase + 1536);
            v16bf c0 = *(const v16bf*)(B1 + bbase);
            v16bf c1 = *(const v16bf*)(B1 + bbase + 512);
            v16bf c2 = *(const v16bf*)(B1 + bbase + 1024);
            v16bf c3 = *(const v16bf*)(B1 + bbase + 1536);
            acc[0][0] = __builtin_amdgcn_wmma_f32_16x16x32_bf16(false, a00, false, b0, (short)0, acc[0][0], false, false);
            acc[1][0] = __builtin_amdgcn_wmma_f32_16x16x32_bf16(false, a01, false, b0, (short)0, acc[1][0], false, false);
            acc[0][1] = __builtin_amdgcn_wmma_f32_16x16x32_bf16(false, a00, false, b1, (short)0, acc[0][1], false, false);
            acc[1][1] = __builtin_amdgcn_wmma_f32_16x16x32_bf16(false, a01, false, b1, (short)0, acc[1][1], false, false);
            acc[0][2] = __builtin_amdgcn_wmma_f32_16x16x32_bf16(false, a00, false, b2, (short)0, acc[0][2], false, false);
            acc[1][2] = __builtin_amdgcn_wmma_f32_16x16x32_bf16(false, a01, false, b2, (short)0, acc[1][2], false, false);
            acc[0][3] = __builtin_amdgcn_wmma_f32_16x16x32_bf16(false, a00, false, b3, (short)0, acc[0][3], false, false);
            acc[1][3] = __builtin_amdgcn_wmma_f32_16x16x32_bf16(false, a01, false, b3, (short)0, acc[1][3], false, false);
            acc[0][0] = __builtin_amdgcn_wmma_f32_16x16x32_bf16(false, a10, false, c0, (short)0, acc[0][0], false, false);
            acc[1][0] = __builtin_amdgcn_wmma_f32_16x16x32_bf16(false, a11, false, c0, (short)0, acc[1][0], false, false);
            acc[0][1] = __builtin_amdgcn_wmma_f32_16x16x32_bf16(false, a10, false, c1, (short)0, acc[0][1], false, false);
            acc[1][1] = __builtin_amdgcn_wmma_f32_16x16x32_bf16(false, a11, false, c1, (short)0, acc[1][1], false, false);
            acc[0][2] = __builtin_amdgcn_wmma_f32_16x16x32_bf16(false, a10, false, c2, (short)0, acc[0][2], false, false);
            acc[1][2] = __builtin_amdgcn_wmma_f32_16x16x32_bf16(false, a11, false, c2, (short)0, acc[1][2], false, false);
            acc[0][3] = __builtin_amdgcn_wmma_f32_16x16x32_bf16(false, a10, false, c3, (short)0, acc[0][3], false, false);
            acc[1][3] = __builtin_amdgcn_wmma_f32_16x16x32_bf16(false, a11, false, c3, (short)0, acc[1][3], false, false);
        }
    } else {
        for (int kt = 0; kt < kt_end; ++kt) {
            const size_t bbase =
                (((size_t)kt * ntiles + (nb >> 4)) << 9) + (size_t)lane * 16;
            v16bf a00 = load_abf(a0p0 + kt * 32);
            v16bf a01 = load_abf(a0p1 + kt * 32);
            v16bf b0 = *(const v16bf*)(B0 + bbase);
            v16bf b1 = *(const v16bf*)(B0 + bbase + 512);
            v16bf b2 = *(const v16bf*)(B0 + bbase + 1024);
            v16bf b3 = *(const v16bf*)(B0 + bbase + 1536);
            acc[0][0] = __builtin_amdgcn_wmma_f32_16x16x32_bf16(false, a00, false, b0, (short)0, acc[0][0], false, false);
            acc[1][0] = __builtin_amdgcn_wmma_f32_16x16x32_bf16(false, a01, false, b0, (short)0, acc[1][0], false, false);
            acc[0][1] = __builtin_amdgcn_wmma_f32_16x16x32_bf16(false, a00, false, b1, (short)0, acc[0][1], false, false);
            acc[1][1] = __builtin_amdgcn_wmma_f32_16x16x32_bf16(false, a01, false, b1, (short)0, acc[1][1], false, false);
            acc[0][2] = __builtin_amdgcn_wmma_f32_16x16x32_bf16(false, a00, false, b2, (short)0, acc[0][2], false, false);
            acc[1][2] = __builtin_amdgcn_wmma_f32_16x16x32_bf16(false, a01, false, b2, (short)0, acc[1][2], false, false);
            acc[0][3] = __builtin_amdgcn_wmma_f32_16x16x32_bf16(false, a00, false, b3, (short)0, acc[0][3], false, false);
            acc[1][3] = __builtin_amdgcn_wmma_f32_16x16x32_bf16(false, a01, false, b3, (short)0, acc[1][3], false, false);
        }
    }

    // store: VGPR r -> row m0 + rt*16 + 8*half + r, col = nb + 16*j + (lane&15)
    #pragma unroll
    for (int rt = 0; rt < 2; ++rt) {
        const int srow = m0 + rt * 16 + half * 8;
        #pragma unroll
        for (int j = 0; j < 4; ++j) {
            int col = nb + j * 16 + r15;
            float bv = (bias != nullptr) ? bias[col] : 0.0f;
            #pragma unroll
            for (int r = 0; r < 8; ++r) {
                int row = srow + r;
                if (row < M) {
                    float v = acc[rt][j][r] + bv;
                    if (relu) v = fmaxf(v, 0.0f);
                    D[(size_t)row * Nout + col] = v;
                }
            }
        }
    }
}

// ---------- BN stats: column sums / sumsq ----------
__global__ void bn_stats_kernel(const float* __restrict__ Y, float* __restrict__ sum,
                                float* __restrict__ sq, int Nn, int F) {
    __shared__ float sh[4][64];
    __shared__ float sh2[4][64];
    int c63 = threadIdx.x & 63;
    int c = blockIdx.x * 64 + c63;
    int g = threadIdx.x >> 6;
    int stride = gridDim.y * 4;
    float s = 0.f, s2 = 0.f;
    for (int r = blockIdx.y * 4 + g; r < Nn; r += stride) {
        float v = Y[(size_t)r * F + c];
        s += v; s2 += v * v;
    }
    sh[g][c63] = s; sh2[g][c63] = s2;
    __syncthreads();
    if (g == 0) {
        float S  = sh[0][c63] + sh[1][c63] + sh[2][c63] + sh[3][c63];
        float S2 = sh2[0][c63] + sh2[1][c63] + sh2[2][c63] + sh2[3][c63];
        atomicAdd(&sum[c], S);
        atomicAdd(&sq[c], S2);
    }
}

__global__ void bn_finalize_kernel(const float* __restrict__ sum, const float* __restrict__ sq,
                                   float* __restrict__ mean, float* __restrict__ rstd,
                                   int F, float invN) {
    int c = blockIdx.x * blockDim.x + threadIdx.x;
    if (c >= F) return;
    float m = sum[c] * invN;
    float v = sq[c] * invN - m * m;
    mean[c] = m;
    rstd[c] = rsqrtf(v + 1e-5f);
}

// ---------- out = relu(concat([bn(Y), Xprev])), fp32 + optional bf16 shadow ----------
__global__ void bn_relu_concat_kernel(const float* __restrict__ Y,
                                      const float* __restrict__ mean, const float* __restrict__ rstd,
                                      const float* __restrict__ gam, const float* __restrict__ bet,
                                      const float* __restrict__ Xprev, float* __restrict__ Xout,
                                      bf16_t* __restrict__ XoutBf,
                                      unsigned Nn, unsigned F, unsigned Fp) {
    unsigned Ftot = F + Fp;
    unsigned t = blockIdx.x * blockDim.x + threadIdx.x;
    unsigned total = Nn * Ftot;
    if (t >= total) return;
    unsigned r = t / Ftot;
    unsigned c = t % Ftot;
    float v;
    if (c < F) {
        float y = Y[(size_t)r * F + c];
        v = (y - mean[c]) * rstd[c] * gam[c] + bet[c];
    } else {
        v = Xprev[(size_t)r * Fp + (c - F)];
    }
    float rv = fmaxf(v, 0.0f);
    Xout[t] = rv;
    if (XoutBf) XoutBf[t] = f2bf(rv);
}

// ---------- segment pooling (add + max); x3 >= 0 so uint-bit max is valid ----------
__global__ void pool_scatter_kernel(const float* __restrict__ x3, const int* __restrict__ batch,
                                    float* __restrict__ h, unsigned Nn, unsigned Dd) {
    unsigned t = blockIdx.x * blockDim.x + threadIdx.x;
    unsigned total = Nn * Dd;
    if (t >= total) return;
    unsigned r = t / Dd;
    unsigned c = t % Dd;
    int g = batch[r];
    float v = x3[t];
    atomicAdd(&h[(size_t)g * 4608 + c], v);
    atomicMax((unsigned int*)&h[(size_t)g * 4608 + 1536 + c], __float_as_uint(v));
}

__global__ void pool_count_kernel(const int* __restrict__ batch, float* __restrict__ cnt, int Nn) {
    int t = blockIdx.x * blockDim.x + threadIdx.x;
    if (t >= Nn) return;
    atomicAdd(&cnt[batch[t]], 1.0f);
}

__global__ void pool_mean_kernel(float* __restrict__ h, const float* __restrict__ cnt,
                                 unsigned G, unsigned Dd) {
    unsigned t = blockIdx.x * blockDim.x + threadIdx.x;
    unsigned total = G * Dd;
    if (t >= total) return;
    unsigned g = t / Dd;
    unsigned c = t % Dd;
    h[(size_t)g * 4608 + 3072 + c] = h[(size_t)g * 4608 + c] / fmaxf(cnt[g], 1.0f);
}

// ---------- final 2-class head + log_softmax ----------
__global__ void mlp_final_kernel(const float* __restrict__ h2, const float* __restrict__ Wm3,
                                 const float* __restrict__ bm3, float* __restrict__ out) {
    __shared__ float r0[256];
    __shared__ float r1[256];
    int g = blockIdx.x;
    int tid = threadIdx.x;
    float s0 = 0.f, s1 = 0.f;
    for (int c = tid; c < 2304; c += 256) {
        float v = h2[(size_t)g * 2304 + c];
        s0 += v * Wm3[c * 2 + 0];
        s1 += v * Wm3[c * 2 + 1];
    }
    r0[tid] = s0; r1[tid] = s1;
    __syncthreads();
    for (int s = 128; s > 0; s >>= 1) {
        if (tid < s) { r0[tid] += r0[tid + s]; r1[tid] += r1[tid + s]; }
        __syncthreads();
    }
    if (tid == 0) {
        float l0 = r0[0] + bm3[0];
        float l1 = r1[0] + bm3[1];
        float m = fmaxf(l0, l1);
        float lse = m + logf(expf(l0 - m) + expf(l1 - m));
        out[g * 2 + 0] = l0 - lse;
        out[g * 2 + 1] = l1 - lse;
    }
}

// ---------------------------------------------------------------------------
static inline size_t cdiv(size_t a, size_t b) { return (a + b - 1) / b; }

extern "C" void kernel_launch(void* const* d_in, const int* in_sizes, int n_in,
                              void* d_out, int out_size, void* d_ws, size_t ws_size,
                              hipStream_t stream) {
    (void)in_sizes; (void)n_in; (void)out_size; (void)ws_size;
    const int N = 10000, E = 160000, G = 64;
    const int C = 128, H = 256, F2 = 384, F3 = 768, D = 1536, P = 4608, PH = 2304;

    const float* x        = (const float*)d_in[0];
    const int*   esrc     = (const int*)  d_in[1];
    const int*   edst     = (const int*)  d_in[2];
    const float* ew       = (const float*)d_in[3];
    const int*   batch    = (const int*)  d_in[4];
    const float* W1_rel   = (const float*)d_in[5];
    const float* b1_rel   = (const float*)d_in[6];
    const float* W1_root  = (const float*)d_in[7];
    const float* g1       = (const float*)d_in[8];
    const float* be1      = (const float*)d_in[9];
    const float* W2_rel   = (const float*)d_in[10];
    const float* b2_rel   = (const float*)d_in[11];
    const float* W2_root  = (const float*)d_in[12];
    const float* g2       = (const float*)d_in[13];
    const float* be2      = (const float*)d_in[14];
    const float* W3_rel   = (const float*)d_in[15];
    const float* b3_rel   = (const float*)d_in[16];
    const float* W3_root  = (const float*)d_in[17];
    const float* g3       = (const float*)d_in[18];
    const float* be3      = (const float*)d_in[19];
    const float* Wm1      = (const float*)d_in[20];
    const float* bm1      = (const float*)d_in[21];
    const float* Wm2      = (const float*)d_in[22];
    const float* bm2      = (const float*)d_in[23];
    const float* Wm3      = (const float*)d_in[24];
    const float* bm3      = (const float*)d_in[25];

    // ---- carve workspace ----
    char* base = (char*)d_ws;
    size_t off = 0;
    auto carve = [&](size_t bytes) -> void* {
        off = (off + 255) & ~(size_t)255;
        void* p = base + off;
        off += bytes;
        return p;
    };
    bf16_t* wp1r  = (bf16_t*)carve((size_t)C  * H  * 2);
    bf16_t* wp1t  = (bf16_t*)carve((size_t)C  * H  * 2);
    bf16_t* wp2r  = (bf16_t*)carve((size_t)F2 * F2 * 2);
    bf16_t* wp2t  = (bf16_t*)carve((size_t)F2 * F2 * 2);
    bf16_t* wp3r  = (bf16_t*)carve((size_t)F3 * F3 * 2);
    bf16_t* wp3t  = (bf16_t*)carve((size_t)F3 * F3 * 2);
    bf16_t* wpm1  = (bf16_t*)carve((size_t)P  * PH * 2);
    bf16_t* wpm2  = (bf16_t*)carve((size_t)PH * PH * 2);
    float*  agg   = (float*) carve((size_t)N * F3 * 4);   // reused per layer
    bf16_t* aggbf = (bf16_t*)carve((size_t)N * F3 * 2);   // reused per layer
    float*  ybuf  = (float*) carve((size_t)N * F3 * 4);   // reused per layer
    float*  x1    = (float*) carve((size_t)N * F2 * 4);
    bf16_t* x1bf  = (bf16_t*)carve((size_t)N * F2 * 2);
    float*  x2    = (float*) carve((size_t)N * F3 * 4);
    bf16_t* x2bf  = (bf16_t*)carve((size_t)N * F3 * 2);
    float*  x3    = (float*) carve((size_t)N * D  * 4);
    bf16_t* xbf   = (bf16_t*)carve((size_t)N * C  * 2);
    float*  ssum  = (float*) carve((size_t)F3 * 4);
    float*  ssq   = (float*) carve((size_t)F3 * 4);
    float*  smean = (float*) carve((size_t)F3 * 4);
    float*  srstd = (float*) carve((size_t)F3 * 4);
    float*  hbuf  = (float*) carve((size_t)G * P * 4);
    bf16_t* hbf   = (bf16_t*)carve((size_t)G * P * 2);
    float*  cnt   = (float*) carve((size_t)G * 4);
    float*  h1    = (float*) carve((size_t)G * PH * 4);
    bf16_t* h1bf  = (bf16_t*)carve((size_t)G * PH * 2);
    float*  h2    = (float*) carve((size_t)G * PH * 4);

    const int TB = 256;
    auto cvt = [&](const float* in, bf16_t* o, size_t n) {
        unsigned n4 = (unsigned)(n >> 2);
        cvt_f32_bf16_kernel<<<(unsigned)cdiv(n4, TB), TB, 0, stream>>>(in, o, n4);
    };

    // ---- pack all weights to bf16 B-fragment layout ----
    auto pack = [&](const float* W, bf16_t* o, int K, int Nout) {
        size_t tot = (size_t)K * Nout;
        pack_w_kernel<<<(unsigned)cdiv(tot, TB), TB, 0, stream>>>(W, o, (unsigned)K, (unsigned)Nout);
    };
    pack(W1_rel, wp1r, C, H);   pack(W1_root, wp1t, C, H);
    pack(W2_rel, wp2r, F2, F2); pack(W2_root, wp2t, F2, F2);
    pack(W3_rel, wp3r, F3, F3); pack(W3_root, wp3t, F3, F3);
    pack(Wm1,   wpm1, P, PH);   pack(Wm2,    wpm2, PH, PH);

    cvt(x, xbf, (size_t)N * C);

    auto run_layer = [&](const float* xin, const bf16_t* xinbf, int Fin, int Fout,
                         const bf16_t* Wr, const bf16_t* Wt, const float* bias,
                         const float* gam, const float* bet,
                         float* xout, bf16_t* xoutbf) {
        hipMemsetAsync(agg, 0, (size_t)N * Fin * 4, stream);
        size_t stot = (size_t)E * (Fin >> 2);
        scatter_edges_kernel<<<(unsigned)cdiv(stot, TB), TB, 0, stream>>>(
            xin, esrc, edst, ew, agg, (unsigned)E, (unsigned)Fin);
        cvt(agg, aggbf, (size_t)N * Fin);
        dim3 ggrid(Fout / 64, (unsigned)cdiv(N, 128));
        gemm_bf16_wmma<<<ggrid, 128, 0, stream>>>(
            aggbf, Wr, xinbf, Wt, bias, ybuf, N, Fin, Fout, 0);
        hipMemsetAsync(ssum, 0, (size_t)Fout * 4, stream);
        hipMemsetAsync(ssq,  0, (size_t)Fout * 4, stream);
        dim3 sgrid(Fout / 64, 32);
        bn_stats_kernel<<<sgrid, 256, 0, stream>>>(ybuf, ssum, ssq, N, Fout);
        bn_finalize_kernel<<<(unsigned)cdiv(Fout, TB), TB, 0, stream>>>(
            ssum, ssq, smean, srstd, Fout, 1.0f / (float)N);
        size_t ctot = (size_t)N * (Fout + Fin);
        bn_relu_concat_kernel<<<(unsigned)cdiv(ctot, TB), TB, 0, stream>>>(
            ybuf, smean, srstd, gam, bet, xin, xout, xoutbf,
            (unsigned)N, (unsigned)Fout, (unsigned)Fin);
    };

    run_layer(x,  xbf,  C,  H,  wp1r, wp1t, b1_rel, g1, be1, x1, x1bf);   // x1: [N,384]
    run_layer(x1, x1bf, F2, F2, wp2r, wp2t, b2_rel, g2, be2, x2, x2bf);   // x2: [N,768]
    run_layer(x2, x2bf, F3, F3, wp3r, wp3t, b3_rel, g3, be3, x3, nullptr);// x3: [N,1536]

    // ---- pooling: h = [add | max | mean] ----
    hipMemsetAsync(hbuf, 0, (size_t)G * P * 4, stream);
    hipMemsetAsync(cnt,  0, (size_t)G * 4, stream);
    size_t ptot = (size_t)N * D;
    pool_scatter_kernel<<<(unsigned)cdiv(ptot, TB), TB, 0, stream>>>(
        x3, batch, hbuf, (unsigned)N, (unsigned)D);
    pool_count_kernel<<<(unsigned)cdiv(N, TB), TB, 0, stream>>>(batch, cnt, N);
    pool_mean_kernel<<<(unsigned)cdiv((size_t)G * D, TB), TB, 0, stream>>>(
        hbuf, cnt, (unsigned)G, (unsigned)D);
    cvt(hbuf, hbf, (size_t)G * P);

    // ---- MLP head ----
    {
        dim3 ggrid(PH / 64, (unsigned)cdiv(G, 128));
        gemm_bf16_wmma<<<ggrid, 128, 0, stream>>>(hbf, wpm1, nullptr, nullptr, bm1, h1, G, P,  PH, 1);
        cvt(h1, h1bf, (size_t)G * PH);
        gemm_bf16_wmma<<<ggrid, 128, 0, stream>>>(h1bf, wpm2, nullptr, nullptr, bm2, h2, G, PH, PH, 1);
    }
    mlp_final_kernel<<<G, 256, 0, stream>>>(h2, Wm3, bm3, (float*)d_out);
}
